// InputLinearizationRnnNoConstraint_59141699666503
// MI455X (gfx1250) — matrix operand order repaired and checked by reading, not compiled, not run on hardware
//
#include <hip/hip_runtime.h>
#include <hip/hip_bf16.h>
#include <math.h>

// ---------------- types ----------------
typedef __attribute__((ext_vector_type(16))) __bf16          v16bf;
typedef __attribute__((ext_vector_type(8)))  float           v8f;
typedef __attribute__((ext_vector_type(4)))  float           v4f;
typedef __attribute__((ext_vector_type(16))) unsigned short  v16u;
typedef __attribute__((ext_vector_type(8)))  unsigned short  v8u;
typedef __attribute__((ext_vector_type(8)))  unsigned int    v8ui;

// ---------------- problem dims ----------------
constexpr int NDc = 32, NEc = 64;
constexpr int BATCH = 512, NSTEP = 1024;
// combined weight shapes (K x N):
constexpr int KG = 160, NG = 64;    // [x;u] -> w     (Wg[k][n] = P[192+n][k])
constexpr int KX = 224, NXW = 128;  // [x;u;w] -> x'  (Wx[k][n] = P[n][k])
constexpr int KY = 224, NY = 64;    // [x';u;w] -> y  (Wy[k][n] = P[128+n][k])
constexpr int WG_BYTES = (KG/32)*(NG/16)*1024;    // 20480
constexpr int WX_BYTES = (KX/32)*(NXW/16)*1024;   // 57344
constexpr int WY_BYTES = (KY/32)*(NY/16)*1024;    // 28672
constexpr int W_TOTAL  = WG_BYTES + WX_BYTES + WY_BYTES; // 106496 B

// ---------------- helpers ----------------
__device__ __forceinline__ unsigned short f2bf(float f) {
  unsigned u = __float_as_uint(f);
  u += 0x7FFFu + ((u >> 16) & 1u);        // round-to-nearest-even
  return (unsigned short)(u >> 16);
}

// two floats -> packed bf16 pair (v_cvt_pk_bf16_f32 when available)
__device__ __forceinline__ unsigned int pk2(float lo, float hi) {
#if __has_builtin(__builtin_amdgcn_cvt_pk_bf16_f32)
  auto p = __builtin_amdgcn_cvt_pk_bf16_f32(lo, hi);
  unsigned int w; __builtin_memcpy(&w, &p, sizeof(w));
  return w;
#else
  return (unsigned int)f2bf(lo) | ((unsigned int)f2bf(hi) << 16);
#endif
}

// tanh via hardware transcendentals: v_exp_f32 + v_rcp_f32.
// Saturates to +/-1 for large |x|, NaN propagates.
__device__ __forceinline__ float fast_tanh(float x) {
  float e = __builtin_amdgcn_exp2f(x * 2.8853900817779268f); // 2*log2(e)
  return 1.0f - 2.0f * __builtin_amdgcn_rcpf(e + 1.0f);
}

__device__ __forceinline__ v16u cat8(v8u lo, v8u hi) {
  return __builtin_shufflevector(lo, hi, 0,1,2,3,4,5,6,7,8,9,10,11,12,13,14,15);
}

// B-fragment: block (kt,nt) of a K x (NT*16) matrix, 512 ushorts per block,
// lane L owns 32 contiguous bytes (pre-swizzled by build_weights).
__device__ __forceinline__ v16u ldB(const unsigned short* base, int kt, int nt,
                                    int NT, int lane) {
  const unsigned short* p = base + ((kt * NT + nt) << 9) + lane * 16;
  v8u lo = *(const v8u*)p;
  v8u hi = *(const v8u*)(p + 8);
  return cat8(lo, hi);
}

// A-fragment (16x32 bf16) from a row-major [16 x roww] bf16 LDS buffer.
// ISA layout: lanes 0-15 -> row M=lane, K {k0..k0+7, k0+16..k0+23};
//             lanes 16-31 -> row M=lane-16, K {k0+8..15, k0+24..31}.
__device__ __forceinline__ v16u ldA(const unsigned short* buf, int roww, int k0,
                                    int lane) {
  int half = lane >> 4, m = lane & 15;
  const unsigned short* p = buf + m * roww + k0 + half * 8;
  v8u lo = *(const v8u*)p;
  v8u hi = *(const v8u*)(p + 16);
  return cat8(lo, hi);
}

// ---------------- P(i,j) of the assembled plant ----------------
__device__ __forceinline__ float Pval(int i, int j,
    const float* Al, const float* Bl, const float* Cl, const float* Dl,
    const float* B2, const float* D2, const float* th) {
  // P = S_s + (S_l @ theta) with column permutation from S_r (swap first two
  // 64-wide column blocks).
  int pj = (j < 64) ? j + 64 : (j < 128 ? j - 64 : j);
  float t;
  if (i < 64) {
    t = 0.f;
    for (int c = 0; c < 16; ++c) t += B2[i * 16 + c] * th[(64 + c) * 224 + pj];
  } else if (i < 128) {
    t = th[(i - 64) * 224 + pj];
  } else if (i < 192) {
    t = 0.f; int r = i - 128;
    for (int c = 0; c < 16; ++c) t += D2[r * 16 + c] * th[(64 + c) * 224 + pj];
  } else {
    t = th[(i - 112) * 224 + pj];
  }
  float s = 0.f;
  if (i < 64) {
    if (j < 64) s = Al[i * 64 + j];
    else if (j >= 128 && j < 160) s = Bl[i * 32 + (j - 128)];
  } else if (i >= 128 && i < 192) {
    int r = i - 128;
    if (j < 64) s = Cl[r * 64 + j];
    else if (j >= 128 && j < 160) s = Dl[r * 32 + (j - 128)];
  }
  return s + t;
}

// ---------------- kernel 1: build bf16 B-fragment-packed weights ----------------
__global__ void build_weights(const float* __restrict__ Al, const float* __restrict__ Bl,
                              const float* __restrict__ Cl, const float* __restrict__ Dl,
                              const float* __restrict__ B2, const float* __restrict__ D2,
                              const float* __restrict__ th, unsigned short* __restrict__ out) {
  constexpr int EG = KG * NG, EX = KX * NXW, EY = KY * NY;
  for (int idx = threadIdx.x; idx < EG + EX + EY; idx += blockDim.x) {
    int k, n, rowbase, NT, uoff;
    if (idx < EG)            { k = idx / NG;  n = idx % NG;  rowbase = 192; NT = NG / 16;  uoff = 0; }
    else if (idx < EG + EX)  { int l = idx - EG;      k = l / NXW; n = l % NXW; rowbase = 0;   NT = NXW / 16; uoff = WG_BYTES / 2; }
    else                     { int l = idx - EG - EX; k = l / NY;  n = l % NY;  rowbase = 128; NT = NY / 16;  uoff = (WG_BYTES + WX_BYTES) / 2; }
    float v = Pval(rowbase + n, k, Al, Bl, Cl, Dl, B2, D2, th);
    // B-fragment swizzle: lanes 0-15 hold K {0..7,16..23} of col n, lanes 16-31 K {8..15,24..31}
    int kt = k >> 5, kr = k & 31, nt = n >> 4, nr = n & 15;
    int lane = nr + (((kr >> 3) & 1) << 4);
    int pos  = (kr & 7) | ((kr >> 4) << 3);
    out[uoff + ((kt * NT + nt) << 9) + lane * 16 + pos] = f2bf(v);
  }
}

// ---------------- kernel 2: recurrent scan ----------------
__global__ __launch_bounds__(32, 1)
void rnn_scan(const float* __restrict__ xp, const unsigned short* __restrict__ wsrc,
              float* __restrict__ out) {
  __shared__ __align__(16) unsigned short wlds[W_TOTAL / 2]; // 104 KB weights
  __shared__ __align__(16) unsigned short xbuf[16 * 128];    // x' C->A staging
  __shared__ __align__(16) unsigned short wbuf[16 * 64];     // w  C->A staging

  const int lane = threadIdx.x;
  { // one-time weight image copy to LDS (320 KB/WGP makes this free)
    const uint4* s = (const uint4*)wsrc;
    uint4* d = (uint4*)wlds;
    for (int i = lane; i < W_TOTAL / 16; i += 32) d[i] = s[i];
  }
  __syncthreads();
  const unsigned short* wg = wlds;
  const unsigned short* wx = wlds + WG_BYTES / 2;
  const unsigned short* wy = wlds + (WG_BYTES + WX_BYTES) / 2;

  const int half = lane >> 4, m = lane & 15;
  const int rowBase = blockIdx.x * 16;
  const size_t brow = (size_t)(rowBase + m);   // this lane's A-matrix row

  v16u xf[4];                                  // x state as 4 bf16 A-fragments
#pragma unroll
  for (int q = 0; q < 4; ++q) xf[q] = (v16u){};
  v8f cx[8];                                   // x' accumulators (live past loop)
#pragma unroll
  for (int q = 0; q < 8; ++q) cx[q] = (v8f){};

  for (int t = 0; t < NSTEP; ++t) {
    // ---- u A-fragment straight from global (coalesced NT b128 loads) ----
    const v4f* up = (const v4f*)(xp + (brow * NSTEP + t) * NDc);
    v4f a0 = __builtin_nontemporal_load(up + half * 2 + 0);
    v4f a1 = __builtin_nontemporal_load(up + half * 2 + 1);
    v4f b0 = __builtin_nontemporal_load(up + half * 2 + 4);
    v4f b1 = __builtin_nontemporal_load(up + half * 2 + 5);
    v8ui up32;
    up32[0] = pk2(a0[0], a0[1]); up32[1] = pk2(a0[2], a0[3]);
    up32[2] = pk2(a1[0], a1[1]); up32[3] = pk2(a1[2], a1[3]);
    up32[4] = pk2(b0[0], b0[1]); up32[5] = pk2(b0[2], b0[3]);
    up32[6] = pk2(b1[0], b1[1]); up32[7] = pk2(b1[2], b1[3]);
    v16u uf = __builtin_bit_cast(v16u, up32);
    if (t + 1 < NSTEP)  // next step's 128B row -> global_prefetch_b8
      __builtin_prefetch(xp + (brow * NSTEP + (t + 1)) * NDc, 0, 1);

    // ---- stage 1: w = tanh([x,u] @ Wg)  (5 K-tiles x 4 N-tiles) ----
    v8f cw[4];
#pragma unroll
    for (int nt = 0; nt < 4; ++nt) cw[nt] = (v8f){};
#pragma unroll
    for (int kt = 0; kt < 5; ++kt) {
      v16bf A = __builtin_bit_cast(v16bf, (kt < 4) ? xf[kt] : uf);
#pragma unroll
      for (int nt = 0; nt < 4; ++nt) {
        v16bf B = __builtin_bit_cast(v16bf, ldB(wg, kt, nt, 4, lane));
        cw[nt] = __builtin_amdgcn_wmma_f32_16x16x32_bf16(
            false, A, false, B, (short)0, cw[nt], false, false);
      }
    }
    // tanh (v_exp/v_rcp), C-layout -> LDS (same-wave LDS ops are in-order)
#pragma unroll
    for (int nt = 0; nt < 4; ++nt) {
      int col = nt * 16 + m;
#pragma unroll
      for (int r = 0; r < 8; ++r)
        wbuf[(half * 8 + r) * 64 + col] = f2bf(fast_tanh(cw[nt][r]));
    }
    v16u wf0 = ldA(wbuf, 64, 0, lane);
    v16u wf1 = ldA(wbuf, 64, 32, lane);

    // ---- stage 2: x' = [x,u,w] @ Wx  (7 x 8) ----
#pragma unroll
    for (int nt = 0; nt < 8; ++nt) cx[nt] = (v8f){};
#pragma unroll
    for (int kt = 0; kt < 7; ++kt) {
      v16u af = (kt < 4) ? xf[kt] : (kt == 4 ? uf : (kt == 5 ? wf0 : wf1));
      v16bf A = __builtin_bit_cast(v16bf, af);
#pragma unroll
      for (int nt = 0; nt < 8; ++nt) {
        v16bf B = __builtin_bit_cast(v16bf, ldB(wx, kt, nt, 8, lane));
        cx[nt] = __builtin_amdgcn_wmma_f32_16x16x32_bf16(
            false, A, false, B, (short)0, cx[nt], false, false);
      }
    }
    // x' C-layout -> LDS -> reload as next-state A-fragments
#pragma unroll
    for (int nt = 0; nt < 8; ++nt) {
      int col = nt * 16 + m;
#pragma unroll
      for (int r = 0; r < 8; ++r)
        xbuf[(half * 8 + r) * 128 + col] = f2bf(cx[nt][r]);
    }
#pragma unroll
    for (int q = 0; q < 4; ++q) xf[q] = ldA(xbuf, 128, q * 32, lane);

    // ---- stage 3: y = [x',u,w] @ Wy  (7 x 4) ----
    v8f cy[4];
#pragma unroll
    for (int nt = 0; nt < 4; ++nt) cy[nt] = (v8f){};
#pragma unroll
    for (int kt = 0; kt < 7; ++kt) {
      v16u af = (kt < 4) ? xf[kt] : (kt == 4 ? uf : (kt == 5 ? wf0 : wf1));
      v16bf A = __builtin_bit_cast(v16bf, af);
#pragma unroll
      for (int nt = 0; nt < 4; ++nt) {
        v16bf B = __builtin_bit_cast(v16bf, ldB(wy, kt, nt, 4, lane));
        cy[nt] = __builtin_amdgcn_wmma_f32_16x16x32_bf16(
            false, A, false, B, (short)0, cy[nt], false, false);
      }
    }
    // y store: (b, t, n) fp32, stream-once -> nontemporal
#pragma unroll
    for (int nt = 0; nt < 4; ++nt) {
      int col = nt * 16 + m;
#pragma unroll
      for (int r = 0; r < 8; ++r) {
        size_t b = (size_t)(rowBase + half * 8 + r);
        __builtin_nontemporal_store(cy[nt][r],
            &out[(b * NSTEP + t) * (size_t)NEc + col]);
      }
    }
  }

  // ---- x_final: cols 0..63 -> out1, 64..127 -> out2 ----
  const size_t YS = (size_t)BATCH * NSTEP * NEc;
#pragma unroll
  for (int nt = 0; nt < 8; ++nt) {
    int col = nt * 16 + m;
#pragma unroll
    for (int r = 0; r < 8; ++r) {
      size_t b = (size_t)(rowBase + half * 8 + r);
      float v = cx[nt][r];
      if (col < 64) out[YS + b * 64 + col] = v;
      else          out[YS + (size_t)BATCH * 64 + b * 64 + (col - 64)] = v;
    }
  }
}

// ---------------- launch ----------------
extern "C" void kernel_launch(void* const* d_in, const int* in_sizes, int n_in,
                              void* d_out, int out_size, void* d_ws, size_t ws_size,
                              hipStream_t stream) {
  (void)in_sizes; (void)n_in; (void)out_size; (void)ws_size;
  const float* xp = (const float*)d_in[0];
  const float* Al = (const float*)d_in[1];
  const float* Bl = (const float*)d_in[2];
  const float* Cl = (const float*)d_in[3];
  const float* Dl = (const float*)d_in[4];
  const float* B2 = (const float*)d_in[5];
  const float* D2 = (const float*)d_in[6];
  const float* th = (const float*)d_in[7];
  unsigned short* wpack = (unsigned short*)d_ws;   // 104 KB used

  build_weights<<<1, 256, 0, stream>>>(Al, Bl, Cl, Dl, B2, D2, th, wpack);
  rnn_scan<<<32, 32, 0, stream>>>(xp, wpack, (float*)d_out);
}
